// CausalSelfAttention_47725676593826
// MI455X (gfx1250) — matrix operand order repaired and checked by reading, not compiled, not run on hardware
//
#include <hip/hip_runtime.h>
#include <hip/hip_bf16.h>

typedef __bf16 bf16;
typedef __attribute__((ext_vector_type(16))) __bf16 v16bf;
typedef __attribute__((ext_vector_type(8)))  __bf16 v8bf;
typedef __attribute__((ext_vector_type(8)))  float  v8f;
typedef __attribute__((ext_vector_type(4)))  int    v4i;

#define BB   2
#define SS   2048
#define DD   2048
#define HH   16
#define KVHN 4
#define HD   128
#define KVD  (KVHN * HD)   // 512
#define MX   (BB * SS)     // 4096

#define AS1 __attribute__((address_space(1)))
#define AS3 __attribute__((address_space(3)))

// ---------- bf16 helpers (bit-exact RNE) ----------
__device__ __forceinline__ bf16 f2bf(float f) {
  unsigned u = __builtin_bit_cast(unsigned, f);
  unsigned r = u + 0x7FFFu + ((u >> 16) & 1u);
  unsigned short h = (unsigned short)(r >> 16);
  return __builtin_bit_cast(bf16, h);
}
__device__ __forceinline__ float bf2f(bf16 b) {
  unsigned short s = __builtin_bit_cast(unsigned short, b);
  unsigned u = ((unsigned)s) << 16;
  return __builtin_bit_cast(float, u);
}

// ---------- async global -> LDS copy (ASYNCcnt path, verified builtin) ----------
__device__ __forceinline__ void async_copy_b128(void* lds, const void* g) {
#if __has_builtin(__builtin_amdgcn_global_load_async_to_lds_b128)
  __builtin_amdgcn_global_load_async_to_lds_b128((AS1 v4i*)g, (AS3 v4i*)lds, 0, 0);
#else
  unsigned loff = (unsigned)(size_t)(AS3 char*)lds;
  unsigned long long ga = (unsigned long long)(size_t)g;
  asm volatile("global_load_async_to_lds_b128 %0, %1, off" :: "v"(loff), "v"(ga) : "memory");
#endif
}
__device__ __forceinline__ void wait_async0() {
#if __has_builtin(__builtin_amdgcn_s_wait_asynccnt)
  __builtin_amdgcn_s_wait_asynccnt(0);
#else
  asm volatile("s_wait_asynccnt 0x0" ::: "memory");
#endif
}

// ---------- WMMA fragment loaders (CDNA5 ISA 7.12.2 layouts, wave32) ----------
// A (16x32, 16-bit): lane L holds row L&15; halves = K at ks..ks+7 and ks+16..ks+23, ks=8*(L>>4)
__device__ __forceinline__ v16bf load_fragA(const bf16* base, int ld) {
  int lane = threadIdx.x & 31;
  const bf16* p = base + (size_t)(lane & 15) * ld + ((lane >> 4) * 8);
  v8bf a = *(const v8bf*)p;
  v8bf b = *(const v8bf*)(p + 16);
  return __builtin_shufflevector(a, b, 0,1,2,3,4,5,6,7,8,9,10,11,12,13,14,15);
}
// B (32x16, 16-bit): lane L holds col L&15; halves = K at ks..ks+7 and ks+16..ks+23, ks=8*(L>>4)
__device__ __forceinline__ v16bf load_fragB(const bf16* base, int ld) {
  int lane = threadIdx.x & 31;
  const bf16* p = base + (size_t)(lane & 15) * ld + ((lane >> 4) * 8);
  v8bf a = *(const v8bf*)p;
  v8bf b = *(const v8bf*)(p + 16);
  return __builtin_shufflevector(a, b, 0,1,2,3,4,5,6,7,8,9,10,11,12,13,14,15);
}
__device__ __forceinline__ v8f wmma_bf16(v16bf a, v16bf b, v8f c) {
  return __builtin_amdgcn_wmma_f32_16x16x32_bf16(false, a, false, b, (short)0, c, false, false);
}

// ---------- f32 -> bf16 cast ----------
__global__ void cast_f32_bf16(const float* __restrict__ in, bf16* __restrict__ out, int n) {
  int i = blockIdx.x * blockDim.x + threadIdx.x;
  if (i < n) out[i] = f2bf(in[i]);
}

// ---------- GEMM: C[M,N] = A[M,K](bf16) * Bw[N,K]^T (bf16), f32 accumulate ----------
// 256 threads = 8 waves; block tile 128x128, BK=32, double-buffered LDS filled by
// async global->LDS b128 copies; waves 4(M) x 2(N), wave tile 32x64 = 8 accumulators.
template <bool OUT_BF16>
__global__ void __launch_bounds__(256) gemm_lds(const bf16* __restrict__ A,
                                                const bf16* __restrict__ Bw,
                                                float* __restrict__ Cf,
                                                bf16* __restrict__ Cb,
                                                int M, int N, int K, int tilesN) {
  __shared__ __align__(32) bf16 As[2][128][32];
  __shared__ __align__(32) bf16 Bs[2][128][32];
  int tile = blockIdx.x;
  int tm = tile / tilesN, tn = tile % tilesN;
  int wave = threadIdx.x >> 5;
  int wm = wave & 3, wn = wave >> 2;
  int lane = threadIdx.x & 31;
  int row_blk = tm * 128, col_blk = tn * 128;
  const bf16* Ag = A  + (size_t)row_blk * K;
  const bf16* Bg = Bw + (size_t)col_blk * K;

  // stage filler: 128x32 bf16 tile = 512 x 16B chunks; 2 chunks (A) + 2 chunks (B) / thread
  auto issue_stage = [&](int ks, int buf) {
#pragma unroll
    for (int t = 0; t < 2; ++t) {
      int c = (int)threadIdx.x + t * 256;
      int row = c >> 2, sub = (c & 3) * 8;
      async_copy_b128(&As[buf][row][sub], Ag + (size_t)row * K + ks + sub);
      async_copy_b128(&Bs[buf][row][sub], Bg + (size_t)row * K + ks + sub);
    }
  };

  v8f acc[2][4];
#pragma unroll
  for (int i = 0; i < 2; ++i)
#pragma unroll
    for (int j = 0; j < 4; ++j)
#pragma unroll
      for (int r = 0; r < 8; ++r) acc[i][j][r] = 0.0f;

  issue_stage(0, 0);
  wait_async0();
  __syncthreads();

  int nk = K / 32;
  for (int it = 0; it < nk; ++it) {
    int buf = it & 1;
    if (it + 1 < nk) issue_stage((it + 1) * 32, buf ^ 1);
    v16bf af[2], bfg[4];
#pragma unroll
    for (int i = 0; i < 2; ++i) af[i] = load_fragA(&As[buf][wm * 32 + i * 16][0], 32);
#pragma unroll
    for (int j = 0; j < 4; ++j) bfg[j] = load_fragB(&Bs[buf][wn * 64 + j * 16][0], 32);
#pragma unroll
    for (int i = 0; i < 2; ++i)
#pragma unroll
      for (int j = 0; j < 4; ++j) acc[i][j] = wmma_bf16(af[i], bfg[j], acc[i][j]);
    if (it + 1 < nk) wait_async0();
    __syncthreads();
  }

  int cn = lane & 15, hf = lane >> 4;
#pragma unroll
  for (int i = 0; i < 2; ++i)
#pragma unroll
    for (int j = 0; j < 4; ++j)
#pragma unroll
      for (int r = 0; r < 8; ++r) {
        int row = row_blk + wm * 32 + i * 16 + r + 8 * hf;
        int col = col_blk + wn * 64 + j * 16 + cn;
        size_t idx = (size_t)row * N + col;
        if (OUT_BF16) Cb[idx] = f2bf(acc[i][j][r]);
        else          Cf[idx] = acc[i][j][r];
      }
}

// ---------- RMS-norm + RoPE (+ optional per-head gain): one wave per (b,s,head) ----------
__global__ void qknorm_rope(const bf16* __restrict__ in, bf16* __restrict__ out,
                            const float* __restrict__ gain, int nheads) {
  int wid  = (int)((blockIdx.x * blockDim.x + threadIdx.x) >> 5);
  int lane = threadIdx.x & 31;
  int head = wid % nheads;
  int s    = (wid / nheads) % SS;
  const bf16* p = in  + (size_t)wid * HD;
  bf16*       q = out + (size_t)wid * HD;
  int p0 = lane, p1 = lane + 32;                 // rope pair indices (pair i <-> i+64)
  float x1a = bf2f(p[p0]), x2a = bf2f(p[p0 + 64]);
  float x1b = bf2f(p[p1]), x2b = bf2f(p[p1 + 64]);
  float sq = x1a * x1a + x2a * x2a + x1b * x1b + x2b * x2b;
#pragma unroll
  for (int o = 16; o >= 1; o >>= 1) sq += __shfl_xor(sq, o, 32);
  float inv = rsqrtf(sq * (1.0f / HD) + 1e-6f);
  if (gain) inv *= gain[head];
  const float L2_10K = 13.287712379549449f;      // log2(10000)
  float a0 = (float)s * exp2f(-(float)p0 * (L2_10K / 64.0f));
  float a1 = (float)s * exp2f(-(float)p1 * (L2_10K / 64.0f));
  float c0 = cosf(a0), s0 = sinf(a0);
  float c1 = cosf(a1), s1 = sinf(a1);
  q[p0]      = f2bf(( x1a * c0 + x2a * s0) * inv);
  q[p0 + 64] = f2bf((-x1a * s0 + x2a * c0) * inv);
  q[p1]      = f2bf(( x1b * c1 + x2b * s1) * inv);
  q[p1 + 64] = f2bf((-x1b * s1 + x2b * c1) * inv);
}

// ---------- V transpose: [b][s][kv][d] -> [b][kv][d][s] ----------
__global__ void transpose_v(const bf16* __restrict__ vr, bf16* __restrict__ vt, int n) {
  int i = blockIdx.x * blockDim.x + threadIdx.x;
  if (i >= n) return;
  int d  = i % HD;
  int kv = (i / HD) % KVHN;
  int s  = (i / (HD * KVHN)) % SS;
  int b  = i / (HD * KVHN * SS);
  vt[(((size_t)b * KVHN + kv) * HD + d) * SS + s] = vr[i];
}

// ---------- Flash attention: 8 waves/block, wave = one 16-row Q tile, 32-key blocks.
// K/V tiles double-buffered in LDS via async DMA, shared by all waves in the block. ----
__global__ void __launch_bounds__(256) flash_attn(const bf16* __restrict__ qh,
                                                  const bf16* __restrict__ kh,
                                                  const bf16* __restrict__ vt,
                                                  bf16* __restrict__ y) {
  __shared__ __align__(32) bf16 Ks[2][32][128];     // 32 keys x HD
  __shared__ __align__(32) bf16 Vs[2][128][32];     // HD x 32 keys (from vT)
  __shared__ __align__(32) bf16 ptile[8][16][32];   // per-wave P tile (C-layout -> A-layout)
  const int QB = SS / 128;
  int gid = blockIdx.x;
  int qb = gid % QB;
  int h  = (gid / QB) % HH;
  int b  = gid / (QB * HH);
  int kv = h / (HH / KVHN);
  int wave = threadIdx.x >> 5;
  int lane = threadIdx.x & 31;
  int cn = lane & 15, hf = lane >> 4;
  int q0 = qb * 128 + wave * 16;

  const bf16* khead = kh + (size_t)b * SS * KVD + (size_t)kv * HD;
  const bf16* vhead = vt + ((size_t)b * KVHN + kv) * (size_t)HD * SS;

  // K tile: 32 rows x 256B = 512 x16B chunks; V tile: 128 rows x 64B = 512 chunks.
  auto issue_kv = [&](int ks, int buf) {
#pragma unroll
    for (int t = 0; t < 2; ++t) {
      int c = (int)threadIdx.x + t * 256;
      int kr = c >> 4, ksub = (c & 15) * 8;
      async_copy_b128(&Ks[buf][kr][ksub], khead + (size_t)(ks + kr) * KVD + ksub);
      int vr = c >> 2, vsub = (c & 3) * 8;
      async_copy_b128(&Vs[buf][vr][vsub], vhead + (size_t)vr * SS + ks + vsub);
    }
  };

  const bf16* qbase = qh + ((size_t)(b * SS + q0) * HH + h) * HD;
  v16bf qf[4];
#pragma unroll
  for (int i = 0; i < 4; ++i) qf[i] = load_fragA(qbase + 32 * i, DD);

  v8f o[8];
#pragma unroll
  for (int d = 0; d < 8; ++d)
#pragma unroll
    for (int r = 0; r < 8; ++r) o[d][r] = 0.0f;
  float m[8], l[8];
#pragma unroll
  for (int r = 0; r < 8; ++r) { m[r] = -1e30f; l[r] = 0.0f; }

  const float scale = 0.08838834764831845f;         // 1/sqrt(128)
  int nkb = (qb * 128 + 128) / 32;                  // uniform causal bound for the block

  issue_kv(0, 0);
  wait_async0();
  __syncthreads();

  for (int ib = 0; ib < nkb; ++ib) {
    int kb = ib * 32;
    int buf = ib & 1;
    if (ib + 1 < nkb) issue_kv(kb + 32, buf ^ 1);

    v8f s0 = {0.f,0.f,0.f,0.f,0.f,0.f,0.f,0.f};
    v8f s1 = {0.f,0.f,0.f,0.f,0.f,0.f,0.f,0.f};
#pragma unroll
    for (int i = 0; i < 4; ++i) {
      v16bf b0 = load_fragB(&Ks[buf][0][32 * i],  128);
      v16bf b1 = load_fragB(&Ks[buf][16][32 * i], 128);
      s0 = wmma_bf16(qf[i], b0, s0);
      s1 = wmma_bf16(qf[i], b1, s1);
    }
    float alpha[8];
#pragma unroll
    for (int r = 0; r < 8; ++r) {
      int qrow = q0 + r + 8 * hf;
      int col0 = kb + cn;
      float v0 = (col0      <= qrow) ? s0[r] * scale : -1e30f;
      float v1 = (col0 + 16 <= qrow) ? s1[r] * scale : -1e30f;
      float rm = fmaxf(v0, v1);
#pragma unroll
      for (int off = 8; off >= 1; off >>= 1) rm = fmaxf(rm, __shfl_xor(rm, off, 16));
      float mn = fmaxf(m[r], rm);
      float p0 = __expf(v0 - mn);
      float p1 = __expf(v1 - mn);
      float rs = p0 + p1;
#pragma unroll
      for (int off = 8; off >= 1; off >>= 1) rs += __shfl_xor(rs, off, 16);
      alpha[r] = __expf(m[r] - mn);
      m[r] = mn;
      l[r] = l[r] * alpha[r] + rs;
      ptile[wave][r + 8 * hf][cn]      = f2bf(p0);
      ptile[wave][r + 8 * hf][cn + 16] = f2bf(p1);
    }
#pragma unroll
    for (int d = 0; d < 8; ++d)
#pragma unroll
      for (int r = 0; r < 8; ++r) o[d][r] *= alpha[r];
    __syncthreads();
    v16bf pf = load_fragA(&ptile[wave][0][0], 32);
#pragma unroll
    for (int d = 0; d < 8; ++d) {
      v16bf vf = load_fragB(&Vs[buf][16 * d][0], 32);
      o[d] = wmma_bf16(pf, vf, o[d]);
    }
    if (ib + 1 < nkb) wait_async0();
    __syncthreads();
  }
#pragma unroll
  for (int r = 0; r < 8; ++r) {
    int qrow = q0 + r + 8 * hf;
    float invl = 1.0f / l[r];
    size_t base = ((size_t)(b * SS + qrow) * HH + h) * HD + cn;
#pragma unroll
    for (int d = 0; d < 8; ++d) y[base + 16 * d] = f2bf(o[d][r] * invl);
  }
}

// ---------- host launcher ----------
extern "C" void kernel_launch(void* const* d_in, const int* in_sizes, int n_in,
                              void* d_out, int out_size, void* d_ws, size_t ws_size,
                              hipStream_t stream) {
  const float* x  = (const float*)d_in[0];
  const float* Wq = (const float*)d_in[1];
  const float* Wk = (const float*)d_in[2];
  const float* Wv = (const float*)d_in[3];
  const float* Wp = (const float*)d_in[4];
  const float* qg = (const float*)d_in[5];
  float* out = (float*)d_out;

  char* ws = (char*)d_ws;
  size_t off = 0;
  auto alloc = [&](size_t bytes) -> char* {
    char* p = ws + off;
    off += (bytes + 255) & ~(size_t)255;
    return p;
  };
  bf16* xb   = (bf16*)alloc((size_t)MX  * DD  * 2);
  bf16* wqb  = (bf16*)alloc((size_t)DD  * DD  * 2);
  bf16* wkb  = (bf16*)alloc((size_t)KVD * DD  * 2);
  bf16* wvb  = (bf16*)alloc((size_t)KVD * DD  * 2);
  bf16* wpb  = (bf16*)alloc((size_t)DD  * DD  * 2);
  bf16* qraw = (bf16*)alloc((size_t)MX  * DD  * 2);
  bf16* kraw = (bf16*)alloc((size_t)MX  * KVD * 2);
  bf16* vraw = (bf16*)alloc((size_t)MX  * KVD * 2);
  bf16* qhat = (bf16*)alloc((size_t)MX  * DD  * 2);
  bf16* khat = (bf16*)alloc((size_t)MX  * KVD * 2);
  bf16* vT   = (bf16*)alloc((size_t)MX  * KVD * 2);
  bf16* ybuf = (bf16*)alloc((size_t)MX  * DD  * 2);

  int n;
  n = MX * DD;  cast_f32_bf16<<<(n + 255) / 256, 256, 0, stream>>>(x,  xb,  n);
  n = DD * DD;  cast_f32_bf16<<<(n + 255) / 256, 256, 0, stream>>>(Wq, wqb, n);
  n = KVD * DD; cast_f32_bf16<<<(n + 255) / 256, 256, 0, stream>>>(Wk, wkb, n);
  n = KVD * DD; cast_f32_bf16<<<(n + 255) / 256, 256, 0, stream>>>(Wv, wvb, n);
  n = DD * DD;  cast_f32_bf16<<<(n + 255) / 256, 256, 0, stream>>>(Wp, wpb, n);

  // QKV projections (bf16 out), LDS-staged async-copy GEMM
  gemm_lds<true><<<(MX / 128) * (DD / 128),  256, 0, stream>>>(xb, wqb, nullptr, qraw, MX, DD,  DD, DD / 128);
  gemm_lds<true><<<(MX / 128) * (KVD / 128), 256, 0, stream>>>(xb, wkb, nullptr, kraw, MX, KVD, DD, KVD / 128);
  gemm_lds<true><<<(MX / 128) * (KVD / 128), 256, 0, stream>>>(xb, wvb, nullptr, vraw, MX, KVD, DD, KVD / 128);

  // RMS-norm + RoPE (+ gain on Q)
  qknorm_rope<<<(MX * HH)   / 4, 128, 0, stream>>>(qraw, qhat, qg,      HH);
  qknorm_rope<<<(MX * KVHN) / 4, 128, 0, stream>>>(kraw, khat, nullptr, KVHN);

  // V transpose for contiguous P*V fragments
  n = MX * KVD; transpose_v<<<(n + 255) / 256, 256, 0, stream>>>(vraw, vT, n);

  // Flash attention: one block per (b, h, 128-row q block)
  flash_attn<<<BB * HH * (SS / 128), 256, 0, stream>>>(qhat, khat, vT, ybuf);

  // Output projection (f32 out -> d_out)
  gemm_lds<false><<<(MX / 128) * (DD / 128), 256, 0, stream>>>(ybuf, wpb, out, nullptr, MX, DD, DD, DD / 128);
}
